// AddlnMatmulBlock_36558761623582
// MI455X (gfx1250) — matrix operand order repaired and checked by reading, not compiled, not run on hardware
//
#include <hip/hip_runtime.h>

// ---------- types ----------
typedef __attribute__((ext_vector_type(16))) __bf16 v16bf;
typedef __attribute__((ext_vector_type(8)))  __bf16 v8bf;
typedef __attribute__((ext_vector_type(8)))  float  v8f;

#define R_ROWS 8192   // B*M = 4*2048
#define N_DIM  1024   // LayerNorm / contraction dim (K of GEMM)
#define D_DIM  4096   // output dim
#define LDT    40     // padded LDS row stride (bf16 elems), 80B: 16B-aligned, conflict-free
#define TSZ    (128 * LDT)          // elems per LDS tile array (5120)
#define TSZB   (TSZ * 2)            // bytes per LDS tile array (10240)
#define KSTEPS (N_DIM / 32)

static __device__ inline v16bf cat16(v8bf a, v8bf b) {
  return __builtin_shufflevector(a, b, 0,1,2,3,4,5,6,7,8,9,10,11,12,13,14,15);
}
static __device__ inline v8bf ld8(const __bf16* p) {
  return *reinterpret_cast<const v8bf*>(p);
}

// async global->LDS 16B copy (GVS form: saddr base + 32-bit byte offset)
static __device__ __forceinline__ void async_cp16(uint32_t lds_off,
                                                  const __bf16* base,
                                                  uint32_t byte_off) {
  asm volatile("global_load_async_to_lds_b128 %0, %1, %2"
               :: "v"(lds_off), "v"(byte_off), "s"(base)
               : "memory");
}

// ---------------------------------------------------------------------------
// Kernel 1: transpose w [K=1024, D=4096] f32 -> wT_hi/wT_lo [D, K] bf16 split
// ---------------------------------------------------------------------------
__global__ __launch_bounds__(256) void wsplit_kernel(const float* __restrict__ w,
                                                     __bf16* __restrict__ wThi,
                                                     __bf16* __restrict__ wTlo) {
  __shared__ float tile[32][33];
  const int d0 = blockIdx.x * 32;
  const int k0 = blockIdx.y * 32;
  const int tx = threadIdx.x & 31;
  const int ty = threadIdx.x >> 5;
#pragma unroll
  for (int p = 0; p < 4; ++p) {
    int ky = ty + p * 8;
    tile[ky][tx] = w[(size_t)(k0 + ky) * D_DIM + d0 + tx];   // coalesced read
  }
  __syncthreads();
#pragma unroll
  for (int p = 0; p < 4; ++p) {
    int dy = ty + p * 8;
    float f  = tile[tx][dy];                                  // stride-33: conflict-free
    __bf16 h = (__bf16)f;
    size_t o = (size_t)(d0 + dy) * N_DIM + k0 + tx;           // coalesced write along k
    wThi[o] = h;
    wTlo[o] = (__bf16)(f - (float)h);
  }
}

// ---------------------------------------------------------------------------
// Kernel 2: out_add = x1+x2; LayerNorm stats; ln split into bf16 hi/lo
// ---------------------------------------------------------------------------
__global__ __launch_bounds__(256) void add_ln_kernel(const float4* __restrict__ x1,
                                                     const float4* __restrict__ x2,
                                                     const float*  __restrict__ gamma,
                                                     const float*  __restrict__ beta,
                                                     float4* __restrict__ out_add,
                                                     float*  __restrict__ meanp,
                                                     float*  __restrict__ rstdp,
                                                     __bf16* __restrict__ lnhi,
                                                     __bf16* __restrict__ lnlo) {
  const int row  = blockIdx.x;
  const int t    = threadIdx.x;
  const int lane = t & 31, wid = t >> 5;
  const size_t base4 = (size_t)row * (N_DIM / 4);

  float4 a = x1[base4 + t];
  float4 b = x2[base4 + t];
  float4 v;
  v.x = a.x + b.x; v.y = a.y + b.y; v.z = a.z + b.z; v.w = a.w + b.w;
  out_add[base4 + t] = v;

  float s  = v.x + v.y + v.z + v.w;
  float sq = v.x * v.x + v.y * v.y + v.z * v.z + v.w * v.w;
#pragma unroll
  for (int o = 16; o > 0; o >>= 1) {          // wave32 reduction
    s  += __shfl_xor(s,  o, 32);
    sq += __shfl_xor(sq, o, 32);
  }
  __shared__ float rs[8], rq[8], smean, srstd;
  if (lane == 0) { rs[wid] = s; rq[wid] = sq; }
  __syncthreads();
  if (t == 0) {
    float S = 0.f, Q = 0.f;
#pragma unroll
    for (int i = 0; i < 8; ++i) { S += rs[i]; Q += rq[i]; }
    float mean = S * (1.0f / N_DIM);
    float var  = Q * (1.0f / N_DIM) - mean * mean;
    float rstd = rsqrtf(var + 1e-5f);
    smean = mean; srstd = rstd;
    meanp[row] = mean; rstdp[row] = rstd;
  }
  __syncthreads();
  const float mean = smean, rstd = srstd;

  const size_t eb = (size_t)row * N_DIM + t * 4;
  float vv[4] = {v.x, v.y, v.z, v.w};
  const float4 g4 = reinterpret_cast<const float4*>(gamma)[t];
  const float4 b4 = reinterpret_cast<const float4*>(beta)[t];
  float gg[4] = {g4.x, g4.y, g4.z, g4.w};
  float bb[4] = {b4.x, b4.y, b4.z, b4.w};
#pragma unroll
  for (int j = 0; j < 4; ++j) {
    float ln = (vv[j] - mean) * rstd * gg[j] + bb[j];
    __bf16 h = (__bf16)ln;
    lnhi[eb + j] = h;
    lnlo[eb + j] = (__bf16)(ln - (float)h);
  }
}

// ---------------------------------------------------------------------------
// Kernel 3: GEMM out[8192,4096] = ln * wT^T + bias, split-bf16 WMMA,
// double-buffered LDS filled by async global->LDS copies (ASYNCcnt pipeline).
// Dynamic LDS layout per buffer: Ahi | Alo | Bhi | Blo, each 128*LDT bf16.
// ---------------------------------------------------------------------------
__global__ __launch_bounds__(256) void gemm_kernel(const __bf16* __restrict__ Ahi_g,
                                                   const __bf16* __restrict__ Alo_g,
                                                   const __bf16* __restrict__ Bhi_g,  // wT hi [D,K]
                                                   const __bf16* __restrict__ Blo_g,  // wT lo [D,K]
                                                   const float*  __restrict__ bias,
                                                   float* __restrict__ out) {
  extern __shared__ __align__(16) __bf16 smem[];   // 2 buffers * 4 arrays * TSZ

  const int t    = threadIdx.x;
  const int lane = t & 31, wid = t >> 5;
  const int wm = wid & 3;            // 4 wave-groups along M (32 rows each)
  const int wn = wid >> 2;           // 2 wave-groups along N (64 cols each)
  const int bn = blockIdx.x;         // D/128
  const int bm = blockIdx.y;         // R/128
  const int lhalf = lane >> 4, lmod = lane & 15;

  v8f acc[2][4];
#pragma unroll
  for (int i = 0; i < 2; ++i)
#pragma unroll
    for (int j = 0; j < 4; ++j) acc[i][j] = 0.f;

  const uint32_t lds0 = (uint32_t)(uintptr_t)smem;

  // issue one stage (8 async b128 per thread): chunk c: row=c>>2, kchunk=(c&3)*8
  auto issue_stage = [&](int k0, int parity) {
    const uint32_t lbase = lds0 + (uint32_t)parity * (4u * TSZB);
#pragma unroll
    for (int cc = 0; cc < 2; ++cc) {
      const int c = t + cc * 256;
      const int r = c >> 2, kc = (c & 3) << 3;
      const uint32_t goffA = 2u * ((uint32_t)(bm * 128 + r) * N_DIM + k0 + kc);
      const uint32_t goffB = 2u * ((uint32_t)(bn * 128 + r) * N_DIM + k0 + kc);
      const uint32_t l = lbase + 2u * (uint32_t)(r * LDT + kc);
      async_cp16(l,            Ahi_g, goffA);
      async_cp16(l + 1u*TSZB,  Alo_g, goffA);
      async_cp16(l + 2u*TSZB,  Bhi_g, goffB);
      async_cp16(l + 3u*TSZB,  Blo_g, goffB);
    }
  };

  issue_stage(0, 0);

  for (int ks = 0; ks < KSTEPS; ++ks) {
    if (ks + 1 < KSTEPS) {
      issue_stage((ks + 1) * 32, (ks + 1) & 1);
      // 8 outstanding = next stage only -> current stage has landed (in-order)
      asm volatile("s_wait_asynccnt 0x8" ::: "memory");
    } else {
      asm volatile("s_wait_asynccnt 0x0" ::: "memory");
    }
    __syncthreads();   // stage ks visible to all waves

    const __bf16* base = smem + (size_t)(ks & 1) * (4 * TSZ);
    const __bf16* sAhi = base;
    const __bf16* sAlo = base + 1 * TSZ;
    const __bf16* sBhi = base + 2 * TSZ;
    const __bf16* sBlo = base + 3 * TSZ;

#pragma unroll
    for (int mt = 0; mt < 2; ++mt) {
      // A fragment (16x32 bf16): lane covers K runs [kh*8,+8) and [16+kh*8,+8)
      const int arow = wm * 32 + mt * 16 + lmod;
      const __bf16* ah = &sAhi[arow * LDT + lhalf * 8];
      const __bf16* al = &sAlo[arow * LDT + lhalf * 8];
      v16bf ahi = cat16(ld8(ah), ld8(ah + 16));
      v16bf alo = cat16(ld8(al), ld8(al + 16));
#pragma unroll
      for (int nt = 0; nt < 4; ++nt) {
        // B fragment (32x16 bf16): lane covers K runs [kh*8,+8) and [16+kh*8,+8)
        const int brow = wn * 64 + nt * 16 + lmod;
        const __bf16* bh = &sBhi[brow * LDT + lhalf * 8];
        const __bf16* bl = &sBlo[brow * LDT + lhalf * 8];
        v16bf bhi = cat16(ld8(bh), ld8(bh + 16));
        v16bf blo = cat16(ld8(bl), ld8(bl + 16));
        // split-precision accumulate: hi*hi + hi*lo + lo*hi
        acc[mt][nt] = __builtin_amdgcn_wmma_f32_16x16x32_bf16(
            false, ahi, false, bhi, (short)0, acc[mt][nt], false, false);
        acc[mt][nt] = __builtin_amdgcn_wmma_f32_16x16x32_bf16(
            false, ahi, false, blo, (short)0, acc[mt][nt], false, false);
        acc[mt][nt] = __builtin_amdgcn_wmma_f32_16x16x32_bf16(
            false, alo, false, bhi, (short)0, acc[mt][nt], false, false);
      }
    }
    __syncthreads();   // all waves done reading buf before it is refilled
  }

  // epilogue: D-tile layout -> lane n = lane%16, VGPR v -> m = v + 8*(lane>=16)
#pragma unroll
  for (int mt = 0; mt < 2; ++mt) {
#pragma unroll
    for (int nt = 0; nt < 4; ++nt) {
      const int n  = bn * 128 + wn * 64 + nt * 16 + lmod;
      const float bv = bias[n];
      const size_t mbase = (size_t)(bm * 128 + wm * 32 + mt * 16 + lhalf * 8);
#pragma unroll
      for (int v = 0; v < 8; ++v) {
        out[(mbase + v) * D_DIM + n] = acc[mt][nt][v] + bv;
      }
    }
  }
}

// ---------------------------------------------------------------------------
extern "C" void kernel_launch(void* const* d_in, const int* in_sizes, int n_in,
                              void* d_out, int out_size, void* d_ws, size_t ws_size,
                              hipStream_t stream) {
  (void)in_sizes; (void)n_in; (void)out_size; (void)ws_size;
  const float* x1    = (const float*)d_in[0];
  const float* x2    = (const float*)d_in[1];
  const float* w     = (const float*)d_in[2];
  const float* bias  = (const float*)d_in[3];
  const float* gamma = (const float*)d_in[4];
  const float* beta  = (const float*)d_in[5];

  // d_out = concat(out_add [R,N], mean [R], rstd [R], out [R,D])
  float* out_add = (float*)d_out;
  float* meanp   = out_add + (size_t)R_ROWS * N_DIM;
  float* rstdp   = meanp + R_ROWS;
  float* outp    = rstdp + R_ROWS;

  // workspace: ln hi/lo [R,N] bf16, wT hi/lo [D,K] bf16  (48 MB total)
  __bf16* lnhi = (__bf16*)d_ws;
  __bf16* lnlo = lnhi + (size_t)R_ROWS * N_DIM;
  __bf16* wThi = lnlo + (size_t)R_ROWS * N_DIM;
  __bf16* wTlo = wThi + (size_t)D_DIM * N_DIM;

  const int dyn_lds = 2 * 4 * TSZB;   // 81920 B: double-buffered tile set
  (void)hipFuncSetAttribute(reinterpret_cast<const void*>(gemm_kernel),
                            hipFuncAttributeMaxDynamicSharedMemorySize, dyn_lds);

  wsplit_kernel<<<dim3(D_DIM / 32, N_DIM / 32), 256, 0, stream>>>(w, wThi, wTlo);
  add_ln_kernel<<<R_ROWS, 256, 0, stream>>>((const float4*)x1, (const float4*)x2,
                                            gamma, beta, (float4*)out_add,
                                            meanp, rstdp, lnhi, lnlo);
  gemm_kernel<<<dim3(D_DIM / 128, R_ROWS / 128), 256, dyn_lds, stream>>>(
      lnhi, lnlo, wThi, wTlo, bias, outp);
}
